// SigTKAN_77644418777104
// MI455X (gfx1250) — matrix operand order repaired and checked by reading, not compiled, not run on hardware
//
#include <hip/hip_runtime.h>
#include <hip/hip_bf16.h>
#include <stdint.h>

// ---------------------------------------------------------------------------
// SigTKAN forward for MI455X (gfx1250, wave32, WMMA).
// All GEMMs: v_wmma_f32_16x16x32_bf16, bf16 operands pre-converted/staged so
// the inner loops are wide loads + WMMA (no per-element scalar conversion).
// ---------------------------------------------------------------------------

typedef __attribute__((ext_vector_type(16))) __bf16 v16bf;
typedef __attribute__((ext_vector_type(8)))  __bf16 v8bf;
typedef __attribute__((ext_vector_type(8)))  float  v8f;

#define BB 32
#define TT 256
#define DD 32
#define UU 256
#define SIGD 1056   // D + D*D
#define MM (BB * TT)

static __device__ __forceinline__ __bf16 f2bf(float f) { return (__bf16)f; }

static __device__ __forceinline__ float sigm(float x) {
    return 1.0f / (1.0f + __expf(-x));
}

// branch-free tanh via v_exp_f32; exact +-1 saturation for large |x|
static __device__ __forceinline__ float tanh_fast(float x) {
    return 1.0f - 2.0f / (__expf(2.0f * x) + 1.0f);
}

static __device__ __forceinline__ v16bf join16(v8bf lo, v8bf hi) {
    return __builtin_shufflevector(lo, hi, 0, 1, 2, 3, 4, 5, 6, 7,
                                           8, 9, 10, 11, 12, 13, 14, 15);
}

// 8 cubic B-spline basis values, uniform extended knots over [-1,1]
// knots: g[i] = (i-3)*h - 1, i = 0..11, h = 2/5
static __device__ __forceinline__ void bspl8(float x, float out[8]) {
    const float h = 0.4f;
    float b[11];
#pragma unroll
    for (int i = 0; i < 11; ++i) {
        float g0 = (float)(i - 3) * h - 1.0f;
        float g1 = (float)(i - 2) * h - 1.0f;
        b[i] = (x >= g0 && x < g1) ? 1.0f : 0.0f;
    }
#pragma unroll
    for (int j = 1; j <= 3; ++j) {
        float inv = 1.0f / ((float)j * h);
#pragma unroll
        for (int i = 0; i < 10; ++i) {
            if (i < 11 - j) {
                float gi  = (float)(i - 3) * h - 1.0f;
                float gj1 = (float)(i + j - 2) * h - 1.0f;   // g[i+j+1]
                b[i] = (x - gi) * inv * b[i] + (gj1 - x) * inv * b[i + 1];
            }
        }
    }
#pragma unroll
    for (int i = 0; i < 8; ++i) out[i] = b[i];
}

// ---------------------------------------------------------------------------
// Weight prep: transpose-convert (K,N) f32 -> (N,K) bf16
// ---------------------------------------------------------------------------
__global__ void wt_kernel(const float* __restrict__ W, __bf16* __restrict__ Wt,
                          int K, int N) {
    int idx = blockIdx.x * 256 + threadIdx.x;   // n*K + k, k fastest
    if (idx >= K * N) return;
    int n = idx / K, k = idx - n * K;
    Wt[idx] = f2bf(W[(size_t)k * N + n]);
}

// elementwise f32 -> bf16 (spline weights (N,K,8) are already B-fragment order)
__global__ void cvt_kernel(const float* __restrict__ S, __bf16* __restrict__ D, int n) {
    int idx = blockIdx.x * 256 + threadIdx.x;
    if (idx < n) D[idx] = f2bf(S[idx]);
}

// ---------------------------------------------------------------------------
// Kernel 1: level-2 path signature.  One block/batch, 1024 threads, thread
// (i,j) owns s2[i][j]; cs/dx in LDS; sequential over T.  Writes f32 + bf16.
// sig layout: (B, T, 1056) = [lvl1(32) | lvl2(1024)]
// ---------------------------------------------------------------------------
__global__ void sig_kernel(const float* __restrict__ x,
                           float* __restrict__ sig,
                           __bf16* __restrict__ sigb) {
    const int b   = blockIdx.x;
    const int tid = threadIdx.x;
    const int i   = tid >> 5;
    const int jd  = tid & 31;
    __shared__ float cs[DD];
    __shared__ float dxs[DD];
    const float* xb = x + (size_t)b * TT * DD;
    float*  sb  = sig  + (size_t)b * TT * SIGD;
    __bf16* sbb = sigb + (size_t)b * TT * SIGD;

    for (int k = tid; k < SIGD; k += 1024) { sb[k] = 0.0f; sbb[k] = f2bf(0.0f); }
    if (tid < DD) cs[tid] = 0.0f;
    float s2 = 0.0f;
    __syncthreads();

    for (int t = 1; t < TT; ++t) {
        if (tid < DD) dxs[tid] = xb[(size_t)t * DD + tid] - xb[(size_t)(t - 1) * DD + tid];
        __syncthreads();
        s2 += cs[i] * dxs[jd];
        __syncthreads();
        if (tid < DD) cs[tid] += dxs[tid];
        __syncthreads();
        float*  row  = sb  + (size_t)t * SIGD;
        __bf16* rowb = sbb + (size_t)t * SIGD;
        if (tid < DD) { row[tid] = cs[tid]; rowb[tid] = f2bf(cs[tid]); }
        row[DD + tid]  = s2;
        rowb[DD + tid] = f2bf(s2);
    }
}

// ---------------------------------------------------------------------------
// WMMA fragment layouts (gfx1250, wave32):
// A 16x32 bf16 : lane l -> row l&15 ; element j -> K = h*16 + j as loaded here (contiguous, same as B)
// B 32x16 bf16 : lane l -> col l&15 ; element j -> K = h*16 + j as loaded here (contiguous, same as A)
// C/D 16x16 f32: lane l -> col l&15 ; element j -> row j + 8*h   (h = l>>4)
// ---------------------------------------------------------------------------

// Kernel 2: fused KAN linear  Y = act( silu(X)@Wb + Bspline(X).Ws )
// A-chunks staged cooperatively in LDS (bf16), B from pre-converted bf16.
// grid (M/16, N/64), block 128.  ACT: 0 none, 1 elu.  WB16: also write bf16.
template <int ACT, bool WB16>
__global__ void kan_linear_kernel(const float* __restrict__ X,
                                  const __bf16* __restrict__ Wbt,  // (N,K)
                                  const __bf16* __restrict__ Wsb,  // (N,K*8)
                                  float* __restrict__ Y,
                                  __bf16* __restrict__ Yb,
                                  int M, int K, int N) {
    __shared__ __bf16 As[16 * 32];
    const int tid  = threadIdx.x;           // 0..127
    const int w    = tid >> 5;
    const int lane = tid & 31;
    const int r    = lane & 15;
    const int hsel = lane >> 4;
    const int mbase = blockIdx.x * 16;
    const int nbase = blockIdx.y * 64 + w * 16;
    const int col   = nbase + r;
    const __bf16* wbrow = Wbt + (size_t)col * K;
    const __bf16* wsrow = Wsb + (size_t)col * K * 8;

    v8f acc = {};

    // ---- base path: silu(X) @ Wb ----
#pragma unroll 1
    for (int kc = 0; kc < K; kc += 32) {
#pragma unroll
        for (int e = 0; e < 4; ++e) {
            int idx = tid * 4 + e;          // 0..511 over (row, k)
            int rr = idx >> 5, kk = idx & 31;
            float xv = X[(size_t)(mbase + rr) * K + kc + kk];
            As[rr * 32 + kk] = f2bf(xv / (1.0f + __expf(-xv)));
        }
        __syncthreads();
        v16bf a = join16(*(const v8bf*)&As[r * 32 + hsel * 16],
                         *(const v8bf*)&As[r * 32 + hsel * 16 + 8]);
        v16bf b = join16(*(const v8bf*)&wbrow[kc + hsel * 16],
                         *(const v8bf*)&wbrow[kc + hsel * 16 + 8]);
        acc = __builtin_amdgcn_wmma_f32_16x16x32_bf16(false, a, false, b,
                                                      (short)0, acc, false, false);
        __syncthreads();
    }

    // ---- spline path: expanded K' = 8K; chunk 32 = 4 features x 8 basis ----
#pragma unroll 1
    for (int fb = 0; fb < K; fb += 4) {
        if (tid < 64) {
            int rr = tid >> 2, f = tid & 3;
            float s[8];
            bspl8(X[(size_t)(mbase + rr) * K + fb + f], s);
#pragma unroll
            for (int g = 0; g < 8; ++g) As[rr * 32 + f * 8 + g] = f2bf(s[g]);
        }
        __syncthreads();
        v16bf a = join16(*(const v8bf*)&As[r * 32 + hsel * 16],
                         *(const v8bf*)&As[r * 32 + hsel * 16 + 8]);
        v16bf b = join16(*(const v8bf*)&wsrow[fb * 8 + hsel * 16],
                         *(const v8bf*)&wsrow[fb * 8 + hsel * 16 + 8]);
        acc = __builtin_amdgcn_wmma_f32_16x16x32_bf16(false, a, false, b,
                                                      (short)0, acc, false, false);
        __syncthreads();
    }

#pragma unroll
    for (int j = 0; j < 8; ++j) {
        int row = mbase + j + 8 * hsel;
        float v = acc[j];
        if (ACT == 1) v = (v > 0.0f) ? v : (__expf(v) - 1.0f);   // elu
        Y[(size_t)row * N + col] = v;
        if (WB16) Yb[(size_t)row * N + col] = f2bf(v);
    }
}

// Kernel 3: plain GEMM  Y = act(X@W + bias), bf16 A (M,K) and Wt (N,K).
// ACT: 0 none, 1 sigmoid, 2 tanh.
template <int ACT, bool HASB>
__global__ void gemm_kernel(const __bf16* __restrict__ X,
                            const __bf16* __restrict__ Wt,
                            const float* __restrict__ bias,
                            float* __restrict__ Y,
                            int M, int K, int N) {
    const int w    = threadIdx.x >> 5;
    const int lane = threadIdx.x & 31;
    const int r    = lane & 15;
    const int hsel = lane >> 4;
    const int mbase = blockIdx.x * 16;
    const int nbase = blockIdx.y * 64 + w * 16;
    const int col   = nbase + r;
    const __bf16* xrow = X  + (size_t)(mbase + r) * K;
    const __bf16* wrow = Wt + (size_t)col * K;

    v8f acc = {};
#pragma unroll 2
    for (int kc = 0; kc < K; kc += 32) {
        v16bf a = join16(*(const v8bf*)&xrow[kc + hsel * 16],
                         *(const v8bf*)&xrow[kc + hsel * 16 + 8]);
        v16bf b = join16(*(const v8bf*)&wrow[kc + hsel * 16],
                         *(const v8bf*)&wrow[kc + hsel * 16 + 8]);
        acc = __builtin_amdgcn_wmma_f32_16x16x32_bf16(false, a, false, b,
                                                      (short)0, acc, false, false);
    }
    float bv = HASB ? bias[col] : 0.0f;
#pragma unroll
    for (int j = 0; j < 8; ++j) {
        int row = mbase + j + 8 * hsel;
        float v = acc[j] + bv;
        if (ACT == 1) v = sigm(v);
        if (ACT == 2) v = tanh_fast(v);
        Y[(size_t)row * N + col] = v;
    }
}

// Kernel 4: attn = softmax( tg*tv + skip ) over U, one block per (b,t) row
__global__ void attn_softmax_kernel(const float* __restrict__ tg,
                                    const float* __restrict__ tv,
                                    const float* __restrict__ sk,
                                    float* __restrict__ attn) {
    const int row = blockIdx.x;
    const int u   = threadIdx.x;
    const size_t off = (size_t)row * UU + u;
    float logit = tg[off] * tv[off] + sk[off];
    __shared__ float red[UU];
    red[u] = logit;
    __syncthreads();
    for (int s = 128; s > 0; s >>= 1) {
        if (u < s) red[u] = fmaxf(red[u], red[u + s]);
        __syncthreads();
    }
    float m = red[0];
    __syncthreads();
    float e = __expf(logit - m);
    red[u] = e;
    __syncthreads();
    for (int s = 128; s > 0; s >>= 1) {
        if (u < s) red[u] += red[u + s];
        __syncthreads();
    }
    attn[off] = e / red[0];
}

// Kernel 5: cur(bf16) = attn * cummean_t(kan_out)
__global__ void prefix_cur_kernel(const float* __restrict__ attn,
                                  const float* __restrict__ kan_out,
                                  __bf16* __restrict__ curb) {
    const int b = blockIdx.x;
    const int u = threadIdx.x;
    const size_t base = (size_t)b * TT * UU + u;
    float acc = 0.0f;
    for (int t = 0; t < TT; ++t) {
        size_t o = base + (size_t)t * UU;
        acc += kan_out[o];
        curb[o] = f2bf(attn[o] * (acc / (float)(t + 1)));
    }
}

// ---------------------------------------------------------------------------
// Kernel 6: LSTM scan.  One workgroup, 16 waves (512 threads), wave w owns
// N-tile w; each wave computes BOTH M-tiles (rows 0-15, 16-31) for all 4
// gates, so one set of B fragments feeds two WMMAs.  h kept as bf16 in LDS,
// c in registers.  kc loop NOT unrolled so weight fragments are re-loaded
// from L2 each iteration instead of being hoisted and spilled to scratch.
// ---------------------------------------------------------------------------
__global__ void lstm_scan_kernel(const float* __restrict__ gi,
                                 const float* __restrict__ gf,
                                 const float* __restrict__ gc,
                                 const float* __restrict__ go,
                                 const __bf16* __restrict__ uit,  // (N,K) bf16
                                 const __bf16* __restrict__ uft,
                                 const __bf16* __restrict__ uct,
                                 const __bf16* __restrict__ uot,
                                 float* __restrict__ out) {
    __shared__ __bf16 hs[BB * UU];          // 16 KB
    const int tid  = threadIdx.x;           // 0..511 (16 waves)
    const int w    = tid >> 5;              // N-tile
    const int lane = tid & 31;
    const int r    = lane & 15;
    const int hsel = lane >> 4;
    const int col  = w * 16 + r;
    const __bf16* uir = uit + (size_t)col * UU;
    const __bf16* ufr = uft + (size_t)col * UU;
    const __bf16* ucr = uct + (size_t)col * UU;
    const __bf16* uor = uot + (size_t)col * UU;

    for (int k = tid; k < BB * UU; k += 512) hs[k] = f2bf(0.0f);
    v8f c0 = {}, c1 = {};                    // cell state rows 0-15 / 16-31
    __syncthreads();

    for (int t = 0; t < TT; ++t) {
        v8f m0i = {}, m0f = {}, m0c = {}, m0o = {};
        v8f m1i = {}, m1f = {}, m1c = {}, m1o = {};
#pragma unroll 1
        for (int kc = 0; kc < UU; kc += 32) {
            v16bf a0 = join16(*(const v8bf*)&hs[r * UU + kc + hsel * 16],
                              *(const v8bf*)&hs[r * UU + kc + hsel * 16 + 8]);
            v16bf a1 = join16(*(const v8bf*)&hs[(16 + r) * UU + kc + hsel * 16],
                              *(const v8bf*)&hs[(16 + r) * UU + kc + hsel * 16 + 8]);
            v16bf bi_ = join16(*(const v8bf*)&uir[kc + hsel * 16],
                               *(const v8bf*)&uir[kc + hsel * 16 + 8]);
            v16bf bf_ = join16(*(const v8bf*)&ufr[kc + hsel * 16],
                               *(const v8bf*)&ufr[kc + hsel * 16 + 8]);
            v16bf bc_ = join16(*(const v8bf*)&ucr[kc + hsel * 16],
                               *(const v8bf*)&ucr[kc + hsel * 16 + 8]);
            v16bf bo_ = join16(*(const v8bf*)&uor[kc + hsel * 16],
                               *(const v8bf*)&uor[kc + hsel * 16 + 8]);
            m0i = __builtin_amdgcn_wmma_f32_16x16x32_bf16(false, a0, false, bi_, (short)0, m0i, false, false);
            m1i = __builtin_amdgcn_wmma_f32_16x16x32_bf16(false, a1, false, bi_, (short)0, m1i, false, false);
            m0f = __builtin_amdgcn_wmma_f32_16x16x32_bf16(false, a0, false, bf_, (short)0, m0f, false, false);
            m1f = __builtin_amdgcn_wmma_f32_16x16x32_bf16(false, a1, false, bf_, (short)0, m1f, false, false);
            m0c = __builtin_amdgcn_wmma_f32_16x16x32_bf16(false, a0, false, bc_, (short)0, m0c, false, false);
            m1c = __builtin_amdgcn_wmma_f32_16x16x32_bf16(false, a1, false, bc_, (short)0, m1c, false, false);
            m0o = __builtin_amdgcn_wmma_f32_16x16x32_bf16(false, a0, false, bo_, (short)0, m0o, false, false);
            m1o = __builtin_amdgcn_wmma_f32_16x16x32_bf16(false, a1, false, bo_, (short)0, m1o, false, false);
        }
        __syncthreads();                     // all h reads done before update
#pragma unroll
        for (int j = 0; j < 8; ++j) {
            {   // rows 0..15
                int row = j + 8 * hsel;
                size_t o = ((size_t)row * TT + t) * UU + col;   // (B,T,U)
                float iv = sigm(gi[o] + m0i[j]);
                float fv = sigm(gf[o] + m0f[j]);
                float cv = tanh_fast(gc[o] + m0c[j]);
                float ov = sigm(go[o] + m0o[j]);
                float cn = fv * c0[j] + iv * cv;
                c0[j] = cn;
                float hv = ov * tanh_fast(cn);
                hs[row * UU + col] = f2bf(hv);
                if (t == TT - 1) out[(size_t)row * UU + col] = hv;
            }
            {   // rows 16..31
                int row = 16 + j + 8 * hsel;
                size_t o = ((size_t)row * TT + t) * UU + col;
                float iv = sigm(gi[o] + m1i[j]);
                float fv = sigm(gf[o] + m1f[j]);
                float cv = tanh_fast(gc[o] + m1c[j]);
                float ov = sigm(go[o] + m1o[j]);
                float cn = fv * c1[j] + iv * cv;
                c1[j] = cn;
                float hv = ov * tanh_fast(cn);
                hs[row * UU + col] = f2bf(hv);
                if (t == TT - 1) out[(size_t)row * UU + col] = hv;
            }
        }
        __syncthreads();
    }
}

// ---------------------------------------------------------------------------
// Host-side pipeline
// ---------------------------------------------------------------------------
extern "C" void kernel_launch(void* const* d_in, const int* in_sizes, int n_in,
                              void* d_out, int out_size, void* d_ws, size_t ws_size,
                              hipStream_t stream) {
    (void)in_sizes; (void)n_in; (void)out_size; (void)ws_size;

    const float* x          = (const float*)d_in[0];
    const float* kan_base   = (const float*)d_in[1];
    const float* kan_spline = (const float*)d_in[2];
    const float* wa_base    = (const float*)d_in[3];
    const float* wa_spline  = (const float*)d_in[4];
    const float* wb_base    = (const float*)d_in[5];
    const float* wb_spline  = (const float*)d_in[6];
    const float* wg         = (const float*)d_in[7];
    const float* wv         = (const float*)d_in[8];
    const float* wskip      = (const float*)d_in[9];
    const float* wi         = (const float*)d_in[10];
    const float* bi         = (const float*)d_in[11];
    const float* wf         = (const float*)d_in[12];
    const float* bf         = (const float*)d_in[13];
    const float* wc         = (const float*)d_in[14];
    const float* bc         = (const float*)d_in[15];
    const float* wo         = (const float*)d_in[16];
    const float* bo         = (const float*)d_in[17];
    const float* ui         = (const float*)d_in[18];
    const float* uf         = (const float*)d_in[19];
    const float* uc         = (const float*)d_in[20];
    const float* uo         = (const float*)d_in[21];
    float* out = (float*)d_out;

    // ---- workspace bump allocator (256B aligned) ----
    char* p = (char*)d_ws;
    auto alloc = [&](size_t bytes) -> char* {
        char* r = p;
        p += (bytes + 255) & ~(size_t)255;
        return r;
    };
    const size_t SZ_SIG = (size_t)MM * SIGD;
    const size_t SZ_U   = (size_t)MM * UU;

    float*  SIG  = (float*)alloc(SZ_SIG * 4);
    float*  KOUT = (float*)alloc(SZ_U * 4);
    float*  H1   = (float*)alloc(SZ_U * 4);
    float*  H2   = (float*)alloc(SZ_U * 4);
    float*  TA   = (float*)alloc(SZ_U * 4);
    float*  TB   = (float*)alloc(SZ_U * 4);
    float*  TC   = (float*)alloc(SZ_U * 4);
    __bf16* SIGB = (__bf16*)alloc(SZ_SIG * 2);
    __bf16* H2B  = (__bf16*)alloc(SZ_U * 2);
    __bf16* CURB = (__bf16*)alloc(SZ_U * 2);

    __bf16* kb_t  = (__bf16*)alloc((size_t)DD * UU * 2);
    __bf16* ks_b  = (__bf16*)alloc((size_t)UU * DD * 8 * 2);
    __bf16* wab_t = (__bf16*)alloc((size_t)SIGD * UU * 2);
    __bf16* was_b = (__bf16*)alloc((size_t)UU * SIGD * 8 * 2);
    __bf16* wbb_t = (__bf16*)alloc((size_t)UU * UU * 2);
    __bf16* wbs_b = (__bf16*)alloc((size_t)UU * UU * 8 * 2);
    __bf16* wg_t  = (__bf16*)alloc((size_t)UU * UU * 2);
    __bf16* wv_t  = (__bf16*)alloc((size_t)UU * UU * 2);
    __bf16* wsk_t = (__bf16*)alloc((size_t)SIGD * UU * 2);
    __bf16* wi_t  = (__bf16*)alloc((size_t)UU * UU * 2);
    __bf16* wf_t  = (__bf16*)alloc((size_t)UU * UU * 2);
    __bf16* wc_t  = (__bf16*)alloc((size_t)UU * UU * 2);
    __bf16* wo_t  = (__bf16*)alloc((size_t)UU * UU * 2);
    __bf16* ui_t  = (__bf16*)alloc((size_t)UU * UU * 2);
    __bf16* uf_t  = (__bf16*)alloc((size_t)UU * UU * 2);
    __bf16* uc_t  = (__bf16*)alloc((size_t)UU * UU * 2);
    __bf16* uo_t  = (__bf16*)alloc((size_t)UU * UU * 2);

    auto wtl = [&](const float* W, __bf16* Wt, int K, int N) {
        wt_kernel<<<(K * N + 255) / 256, 256, 0, stream>>>(W, Wt, K, N);
    };
    auto cvl = [&](const float* S, __bf16* D, int n) {
        cvt_kernel<<<(n + 255) / 256, 256, 0, stream>>>(S, D, n);
    };

    // ---- 0) weight prep ----
    wtl(kan_base, kb_t, DD, UU);
    wtl(wa_base,  wab_t, SIGD, UU);
    wtl(wb_base,  wbb_t, UU, UU);
    wtl(wg, wg_t, UU, UU);
    wtl(wv, wv_t, UU, UU);
    wtl(wskip, wsk_t, SIGD, UU);
    wtl(wi, wi_t, UU, UU);  wtl(wf, wf_t, UU, UU);
    wtl(wc, wc_t, UU, UU);  wtl(wo, wo_t, UU, UU);
    wtl(ui, ui_t, UU, UU);  wtl(uf, uf_t, UU, UU);
    wtl(uc, uc_t, UU, UU);  wtl(uo, uo_t, UU, UU);
    cvl(kan_spline, ks_b,  UU * DD * 8);
    cvl(wa_spline,  was_b, UU * SIGD * 8);
    cvl(wb_spline,  wbs_b, UU * UU * 8);

    const dim3 gblk(128);
    const dim3 ggrid(MM / 16, UU / 64);     // 512 x 4

    // ---- 1) path signature (f32 + bf16) ----
    sig_kernel<<<BB, 1024, 0, stream>>>(x, SIG, SIGB);

    // ---- 2) KAN layers ----
    kan_linear_kernel<0, false><<<ggrid, gblk, 0, stream>>>(x,   kb_t,  ks_b,  KOUT, nullptr, MM, DD,   UU);
    kan_linear_kernel<1, false><<<ggrid, gblk, 0, stream>>>(SIG, wab_t, was_b, H1,   nullptr, MM, SIGD, UU);
    kan_linear_kernel<0, true ><<<ggrid, gblk, 0, stream>>>(H1,  wbb_t, wbs_b, H2,   H2B,     MM, UU,   UU);

    // ---- 3) attention inputs ----
    gemm_kernel<1, false><<<ggrid, gblk, 0, stream>>>(H2B,  wg_t,  nullptr, TA, MM, UU,   UU);
    gemm_kernel<0, false><<<ggrid, gblk, 0, stream>>>(H2B,  wv_t,  nullptr, TB, MM, UU,   UU);
    gemm_kernel<0, false><<<ggrid, gblk, 0, stream>>>(SIGB, wsk_t, nullptr, TC, MM, SIGD, UU);

    // ---- 4) attn = softmax(tg*tv + skip) -> H1 (reuse) ----
    attn_softmax_kernel<<<MM, UU, 0, stream>>>(TA, TB, TC, H1);

    // ---- 5) cur(bf16) = attn * cummean(kan_out) ----
    prefix_cur_kernel<<<BB, UU, 0, stream>>>(H1, KOUT, CURB);

    // ---- 6) gate preactivations (activated) ----
    gemm_kernel<1, true><<<ggrid, gblk, 0, stream>>>(CURB, wi_t, bi, TB,   MM, UU, UU);  // gi
    gemm_kernel<1, true><<<ggrid, gblk, 0, stream>>>(CURB, wf_t, bf, TC,   MM, UU, UU);  // gf
    gemm_kernel<2, true><<<ggrid, gblk, 0, stream>>>(CURB, wc_t, bc, H2,   MM, UU, UU);  // gc
    gemm_kernel<1, true><<<ggrid, gblk, 0, stream>>>(CURB, wo_t, bo, KOUT, MM, UU, UU);  // go

    // ---- 7) sequential LSTM scan -> out (B,U) ----
    lstm_scan_kernel<<<1, 512, 0, stream>>>(TB, TC, H2, KOUT,
                                            ui_t, uf_t, uc_t, uo_t, out);
}
